// GRUModel_43628277793497
// MI455X (gfx1250) — matrix operand order repaired and checked by reading, not compiled, not run on hardware
//
#include <hip/hip_runtime.h>
#include <cstdint>
#include <cstddef>

// ---------------------------------------------------------------------------
// GRU on MI455X (gfx1250, wave32, WMMA):
//  * batch-partitioned persistent recurrence (no inter-WG sync needed)
//  * v_wmma_f32_16x16x32_f16 for both x2h and h2h GEMMs (f32 accumulate)
//  * weights pre-packed to WMMA B-fragment layout (f16) in workspace, served
//    from the 192MB L2 every step
//  * x_t tiles double-buffered in LDS via global_load_async_to_lds_b128
//    (ASYNCcnt) — async-tensor data path of CDNA5
//  * h kept in LDS (f16, padded stride) + per-lane f32 registers
// ---------------------------------------------------------------------------

typedef __attribute__((ext_vector_type(16))) _Float16 v16h;
typedef __attribute__((ext_vector_type(8)))  float    v8f;

#define BATCH 128
#define SEQ   2048
#define INDIM 256
#define HDIM  256
#define G3    768            // 3*HDIM

// LDS layout (single shared array => static LDS offset 0 for async asm)
#define XROWF      260                       // padded x row stride (f32): 1040B == 4 dw mod 64 banks
#define XBUF_BYTES (16 * XROWF * 4)          // 16640 B per x buffer
#define SMH_OFF    (2 * XBUF_BYTES)          // h region starts at 33280
#define HROWH      264                       // padded h row stride (halfs): 528B == 4 dw mod 64 banks
#define SM_BYTES   (SMH_OFF + 16 * HROWH * 2)  // 41728 B total

// ---------------------------------------------------------------------------
// Pack a [768 x 256] f32 weight matrix into f16 WMMA B-fragments.
// Fragment (tile, kk) holds W rows [tile*16, tile*16+16), K [kk*32, kk*32+32).
// Lane l stores column (l&15) of the fragment, K = kk*32 + (l>>4)*8 + j for j<8,
// K = kk*32 + 16 + (l>>4)*8 + (j-8) for j>=8  (matches ISA 16-bit 32x16 B layout, wave32).
// ---------------------------------------------------------------------------
__global__ void gru_pack_w(const float* __restrict__ W, _Float16* __restrict__ P) {
    int i = blockIdx.x * blockDim.x + threadIdx.x;
    if (i >= G3 * INDIM) return;
    int row = i >> 8;          // /256
    int k   = i & 255;
    int tile = row >> 4, lr = row & 15;
    int kk   = k >> 5;
    int chi  = (k >> 3) & 1;
    int j    = (k & 7) | (((k >> 4) & 1) << 3);
    int lane = (chi << 4) | lr;
    P[(((tile << 3) + kk) << 9) + (lane << 4) + j] = (_Float16)W[i];
}

// ----- fragment loaders ----------------------------------------------------

__device__ __forceinline__ v16h load_b_frag(const _Float16* __restrict__ P,
                                            int tile, int kk, int lane) {
    const uint4* p = (const uint4*)(P + (((tile << 3) + kk) << 9) + (lane << 4));
    union { uint4 q[2]; v16h v; } u;
    u.q[0] = p[0];
    u.q[1] = p[1];
    return u.v;
}

// A fragment (16x32 f16) from h tile in LDS. ISA layout:
// lanes 0-15: M=lane, halfs0-7 = K 0..7, halfs8-15 = K 16..23 (rel to kk*32)
// lanes16-31: M=lane-16, K 8..15 / 24..31
__device__ __forceinline__ v16h load_a_h(const unsigned char* smem, int kk, int lane) {
    int base = SMH_OFF + (lane & 15) * (HROWH * 2) + kk * 64 + ((lane >> 4) << 4);
    union { uint4 q[2]; v16h v; } u;
    u.q[0] = *(const uint4*)(smem + base);
    u.q[1] = *(const uint4*)(smem + base + 32);
    return u.v;
}

// A fragment from x_t tile in LDS (stored f32, converted here).
__device__ __forceinline__ v16h load_a_x(const unsigned char* smem, int buf,
                                         int kk, int lane) {
    int kbase = kk * 32 + ((lane >> 4) << 3);
    const float* row = (const float*)(smem + buf * XBUF_BYTES) + (lane & 15) * XROWF + kbase;
    float4 f0 = *(const float4*)(row);
    float4 f1 = *(const float4*)(row + 4);
    float4 f2 = *(const float4*)(row + 16);
    float4 f3 = *(const float4*)(row + 20);
    v16h v;
    v[0]  = (_Float16)f0.x; v[1]  = (_Float16)f0.y; v[2]  = (_Float16)f0.z; v[3]  = (_Float16)f0.w;
    v[4]  = (_Float16)f1.x; v[5]  = (_Float16)f1.y; v[6]  = (_Float16)f1.z; v[7]  = (_Float16)f1.w;
    v[8]  = (_Float16)f2.x; v[9]  = (_Float16)f2.y; v[10] = (_Float16)f2.z; v[11] = (_Float16)f2.w;
    v[12] = (_Float16)f3.x; v[13] = (_Float16)f3.y; v[14] = (_Float16)f3.z; v[15] = (_Float16)f3.w;
    return v;
}

// Async prefetch of x[m0..m0+16, t, :] (16KB) into LDS buffer `buf`.
// 1024 chunks of 16B; 512 threads issue 2 global_load_async_to_lds_b128 each.
__device__ __forceinline__ void prefetch_x(const float* __restrict__ x, int m0,
                                           int t, int buf, int tid) {
    #pragma unroll
    for (int c = tid; c < 1024; c += 512) {
        int row = c >> 6;       // 0..15
        int k16 = c & 63;       // 16B chunk within row
        const float* g = x + ((size_t)(m0 + row) * SEQ + (size_t)t) * INDIM + k16 * 4;
        uint32_t lds = (uint32_t)(buf * XBUF_BYTES + row * (XROWF * 4) + k16 * 16);
        asm volatile("global_load_async_to_lds_b128 %0, %1, off"
                     :: "v"(lds), "v"((unsigned long long)(uintptr_t)g)
                     : "memory");
    }
}

__device__ __forceinline__ float fast_sigmoid(float x) {
    return 1.0f / (1.0f + __expf(-x));
}
__device__ __forceinline__ float fast_tanh(float x) {
    x = fminf(fmaxf(x, -15.0f), 15.0f);
    float e2 = __expf(2.0f * x);
    return (e2 - 1.0f) / (e2 + 1.0f);
}

// ---------------------------------------------------------------------------
// Persistent GRU kernel: 8 blocks x 512 threads (16 wave32s).
// Block b owns batch rows [b*16, b*16+16). Wave w owns h columns [w*16, w*16+16).
// ---------------------------------------------------------------------------
__global__ void __launch_bounds__(512)
gru_recurrence(const float* __restrict__ x,
               const _Float16* __restrict__ WxP,
               const _Float16* __restrict__ WhP,
               const float* __restrict__ bx,
               const float* __restrict__ bh,
               const float* __restrict__ Wfc,
               const float* __restrict__ bfc,
               float* __restrict__ out) {
    __shared__ __align__(16) unsigned char smem[SM_BYTES];

    const int tid  = threadIdx.x;
    const int lane = tid & 31;
    const int w    = tid >> 5;           // wave id 0..15
    const int m0   = blockIdx.x * 16;    // batch-row base
    const int col  = (w << 4) + (lane & 15);   // h column owned by this lane
    const int chi  = lane >> 4;

    // per-lane combined biases (C/D layout: column is constant per lane)
    const float br  = bx[col]       + bh[col];
    const float bz  = bx[256 + col] + bh[256 + col];
    const float bxn = bx[512 + col];
    const float bhn = bh[512 + col];

    // zero h tile in LDS (h0 = 0)
    _Float16* hSm = (_Float16*)(smem + SMH_OFF);
    for (int i = tid; i < 16 * HROWH; i += 512) hSm[i] = (_Float16)0.0f;

    // bring in x_0 asynchronously
    prefetch_x(x, m0, 0, 0, tid);
    asm volatile("s_wait_asynccnt 0" ::: "memory");
    __syncthreads();

    float hprev[8];
    #pragma unroll
    for (int e = 0; e < 8; ++e) hprev[e] = 0.0f;

    for (int t = 0; t < SEQ; ++t) {
        if (t + 1 < SEQ) prefetch_x(x, m0, t + 1, (t + 1) & 1, tid);
        const int buf = t & 1;

        v8f accR  = {};   // sigmoid arg: x@WxR + h@WhR
        v8f accZ  = {};   // sigmoid arg: x@WxZ + h@WhZ
        v8f accXN = {};   // x@WxN  (kept separate: n = tanh(xn + r*hn))
        v8f accHN = {};   // h@WhN

        #pragma unroll
        for (int kk = 0; kk < 8; ++kk) {
            v16h ah = load_a_h(smem, kk, lane);
            v16h b;
            b = load_b_frag(WhP, w,      kk, lane);
            accR  = __builtin_amdgcn_wmma_f32_16x16x32_f16(false, ah, false, b, (short)0, accR,  false, false);
            b = load_b_frag(WhP, 16 + w, kk, lane);
            accZ  = __builtin_amdgcn_wmma_f32_16x16x32_f16(false, ah, false, b, (short)0, accZ,  false, false);
            b = load_b_frag(WhP, 32 + w, kk, lane);
            accHN = __builtin_amdgcn_wmma_f32_16x16x32_f16(false, ah, false, b, (short)0, accHN, false, false);

            v16h axv = load_a_x(smem, buf, kk, lane);
            b = load_b_frag(WxP, w,      kk, lane);
            accR  = __builtin_amdgcn_wmma_f32_16x16x32_f16(false, axv, false, b, (short)0, accR,  false, false);
            b = load_b_frag(WxP, 16 + w, kk, lane);
            accZ  = __builtin_amdgcn_wmma_f32_16x16x32_f16(false, axv, false, b, (short)0, accZ,  false, false);
            b = load_b_frag(WxP, 32 + w, kk, lane);
            accXN = __builtin_amdgcn_wmma_f32_16x16x32_f16(false, axv, false, b, (short)0, accXN, false, false);
        }

        // gate math (per C/D element: VGPR e, M = e + 8*chi, N = col)
        float hnew[8];
        #pragma unroll
        for (int e = 0; e < 8; ++e) {
            float r = fast_sigmoid(accR[e] + br);
            float z = fast_sigmoid(accZ[e] + bz);
            float n = fast_tanh((accXN[e] + bxn) + r * (accHN[e] + bhn));
            hnew[e] = n + z * (hprev[e] - n);
        }

        __syncthreads();   // all waves done reading old h before overwrite

        #pragma unroll
        for (int e = 0; e < 8; ++e) {
            int row = e + (chi << 3);
            hSm[row * HROWH + col] = (_Float16)hnew[e];
            hprev[e] = hnew[e];
        }

        asm volatile("s_wait_asynccnt 0" ::: "memory");  // x_{t+1} in LDS
        __syncthreads();   // h_{t+1-ready} + x ready for everyone
    }

    // ----- final FC: out[m0+r, o] = h[r,:] . Wfc[o,:] + bfc[o] -----
    float* HF = (float*)smem;   // reuse x region as f32 h (16x256)
    #pragma unroll
    for (int e = 0; e < 8; ++e) {
        int row = e + (chi << 3);
        HF[row * 256 + col] = hprev[e];
    }
    __syncthreads();

    if (tid < 32) {
        int r = tid >> 1;
        int o = tid & 1;
        float s = bfc[o];
        const float* wf = Wfc + o * 256;
        for (int k = 0; k < 256; ++k) s += HF[r * 256 + k] * wf[k];
        out[(m0 + r) * 2 + o] = s;
    }
}

// ---------------------------------------------------------------------------
extern "C" void kernel_launch(void* const* d_in, const int* in_sizes, int n_in,
                              void* d_out, int out_size, void* d_ws, size_t ws_size,
                              hipStream_t stream) {
    const float* x   = (const float*)d_in[0];
    const float* Wx  = (const float*)d_in[1];
    const float* bx  = (const float*)d_in[2];
    const float* Wh  = (const float*)d_in[3];
    const float* bh  = (const float*)d_in[4];
    const float* Wfc = (const float*)d_in[5];
    const float* bfc = (const float*)d_in[6];
    float* out = (float*)d_out;

    _Float16* WxP = (_Float16*)d_ws;
    _Float16* WhP = WxP + (size_t)G3 * INDIM;

    const int n = G3 * INDIM;
    gru_pack_w<<<(n + 255) / 256, 256, 0, stream>>>(Wx, WxP);
    gru_pack_w<<<(n + 255) / 256, 256, 0, stream>>>(Wh, WhP);

    gru_recurrence<<<8, 512, 0, stream>>>(x, WxP, WhP, bx, bh, Wfc, bfc, out);
}
